// GCN_LSTM_Learning_adj_Model_22050362097735
// MI455X (gfx1250) — matrix operand order repaired and checked by reading, not compiled, not run on hardware
//
#include <hip/hip_runtime.h>
#include <hip/hip_bf16.h>
#include <math.h>

// ---- problem constants ----
constexpr int Bb  = 64;
constexpr int Nn  = 207;
constexpr int Tt  = 48;
constexpr int Fin = 6;
constexpr int Hh  = 64;
constexpr int G4  = 256;            // 4*H gate width
constexpr int ROWS = Bb * Nn;       // 13248 LSTM rows
constexpr int BUFSTRIDE = 68;       // LDS row stride (floats): 16B aligned, bank-spread

typedef __attribute__((ext_vector_type(16))) __bf16 v16bf;
typedef __attribute__((ext_vector_type(8)))  __bf16 v8bf;
typedef __attribute__((ext_vector_type(8)))  float  v8f;

#define WMMA_BF16(A_, B_, C_) \
  __builtin_amdgcn_wmma_f32_16x16x32_bf16(false, (A_), false, (B_), (short)0, (C_), false, false)

// ---- fragment builders ----

// A-frag from an f32 row (LDS): convert 8+8 floats. Lane half lh:
// K in [kb*32 + 8*lh, +8) -> elems 0..7 ; [kb*32 + 16 + 8*lh, +8) -> elems 8..15
__device__ __forceinline__ void cvt8(const float* p, v16bf& v, int base) {
  float4 q0 = *(const float4*)p;
  float4 q1 = *(const float4*)(p + 4);
  v[base + 0] = (__bf16)q0.x; v[base + 1] = (__bf16)q0.y;
  v[base + 2] = (__bf16)q0.z; v[base + 3] = (__bf16)q0.w;
  v[base + 4] = (__bf16)q1.x; v[base + 5] = (__bf16)q1.y;
  v[base + 6] = (__bf16)q1.z; v[base + 7] = (__bf16)q1.w;
}
__device__ __forceinline__ v16bf loadA_f32row(const float* row, int kb, int lh) {
  v16bf a;
  const float* p = row + kb * 32 + 8 * lh;
  cvt8(p, a, 0);
  cvt8(p + 16, a, 8);
  return a;
}

// A-frag from a bf16 row (global): two 16B chunks, no conversion.
__device__ __forceinline__ v16bf loadA_bf16row(const __bf16* row, int kb, int lh) {
  const __bf16* p = row + kb * 32 + 8 * lh;
  v8bf lo = *(const v8bf*)p;
  v8bf hi = *(const v8bf*)(p + 16);
  return __builtin_shufflevector(lo, hi, 0, 1, 2, 3, 4, 5, 6, 7,
                                 8, 9, 10, 11, 12, 13, 14, 15);
}

// B-frag: two 16B chunks of bf16 from pre-converted weights.
// Layout requirement: Wbf[n*64 + k]; frag = Wbf + n*64 + kb*32 + 8*lh (+16 for elems 8..15).
__device__ __forceinline__ v16bf loadB_bf16(const __bf16* wbase, int n, int kb, int lh) {
  const __bf16* p = wbase + (size_t)n * Hh + kb * 32 + 8 * lh;
  v8bf lo = *(const v8bf*)p;
  v8bf hi = *(const v8bf*)(p + 16);
  return __builtin_shufflevector(lo, hi, 0, 1, 2, 3, 4, 5, 6, 7,
                                 8, 9, 10, 11, 12, 13, 14, 15);
}

// ---------------- weight pre-conversion (once per launch) ----------------
__global__ __launch_bounds__(256) void prep_weights(
    const float* __restrict__ Wih, const float* __restrict__ Whh,
    const float* __restrict__ W2,
    __bf16* __restrict__ wihbf, __bf16* __restrict__ whhbf,
    __bf16* __restrict__ w2t)
{
  const int i = blockIdx.x * 256 + threadIdx.x;
  if (i < G4 * Hh) {             // 16384: row-major bf16 copies
    wihbf[i] = (__bf16)Wih[i];
    whhbf[i] = (__bf16)Whh[i];
  }
  if (i < Hh * Hh) {             // 4096: transposed W2 -> w2t[n*64+k] = W2[k*64+n]
    const int n = i >> 6, k = i & 63;
    w2t[i] = (__bf16)W2[(size_t)k * Hh + n];
  }
}

// ---------------- GCN (two fused layers, one block per (b,t)) ----------------
__global__ __launch_bounds__(256) void gcn_fused(
    const float* __restrict__ x,    // [B,N,T,F]
    const float* __restrict__ W1,   // [F,H]
    const float* __restrict__ b1,   // [H]
    const __bf16* __restrict__ w2t, // [H,H] transposed bf16
    const float* __restrict__ b2,   // [H]
    __bf16* __restrict__ hseq)      // [B,N,T,H] bf16
{
  __shared__ float xs[Nn * Fin];
  __shared__ float buf[208 * BUFSTRIDE];   // 13 row-tiles of 16 (row 207 = zero pad)
  __shared__ float cs[Hh];

  const int bt  = blockIdx.x;
  const int b   = bt / Tt;
  const int t   = bt % Tt;
  const int tid = threadIdx.x;

  // stage x[b,:,t,:]
  const float* xb = x + ((size_t)b * Nn) * Tt * Fin + (size_t)t * Fin;
  for (int i = tid; i < Nn * Fin; i += 256) {
    int n = i / Fin, f = i % Fin;
    xs[i] = xb[(size_t)n * Tt * Fin + f];
  }
  if (tid < Hh) buf[207 * BUFSTRIDE + tid] = 0.0f;  // zero pad row
  __syncthreads();

  // layer 1: Y1 = X @ W1  (K=6, VALU)
  for (int i = tid; i < Nn * Hh; i += 256) {
    int n = i / Hh, h = i % Hh;
    float acc = 0.0f;
#pragma unroll
    for (int f = 0; f < Fin; ++f) acc += xs[n * Fin + f] * W1[f * Hh + h];
    buf[n * BUFSTRIDE + h] = acc;
  }
  __syncthreads();

  const float inv = 1.0f / 208.0f;  // dinv[m]*dinv[n], deg = N+1 everywhere

  // propagate + bias + relu  -> Z1 (in place)
  if (tid < Hh) {
    float s = 0.0f;
    for (int n = 0; n < Nn; ++n) s += buf[n * BUFSTRIDE + tid];
    cs[tid] = s;
  }
  __syncthreads();
  for (int i = tid; i < Nn * Hh; i += 256) {
    int n = i / Hh, h = i % Hh;
    float v = (cs[h] + buf[n * BUFSTRIDE + h]) * inv + b1[h];
    buf[n * BUFSTRIDE + h] = v > 0.0f ? v : 0.0f;
  }
  __syncthreads();

  // layer 2: Y2 = Z1 @ W2 via WMMA (208x64 @ 64x64)
  // 52 output tiles over 8 waves; fixed 7 slots/wave, clamped (no dynamic indexing,
  // waves 4..7 redundantly recompute tile 51 -> everything stays in registers).
  const int lane = tid & 31, wave = tid >> 5;
  const int lh = lane >> 4, m16 = lane & 15;

  v8f acc[7];
#pragma unroll
  for (int q = 0; q < 7; ++q) {
    int tile = wave + q * 8;
    tile = tile < 52 ? tile : 51;
    const int mtq = tile % 13, ntq = tile / 13;
    v8f c = {};
#pragma unroll
    for (int kb = 0; kb < 2; ++kb) {
      v16bf A  = loadA_f32row(&buf[(mtq * 16 + m16) * BUFSTRIDE], kb, lh);
      v16bf Bm = loadB_bf16(w2t, ntq * 16 + m16, kb, lh);
      c = WMMA_BF16(A, Bm, c);
    }
    acc[q] = c;
  }
  __syncthreads();  // everyone done reading Z1

  // dump Y2 tiles back to buf (skip duplicate slots)
#pragma unroll
  for (int q = 0; q < 7; ++q) {
    const int tile = wave + q * 8;
    if (tile < 52) {
      const int mtq = tile % 13, ntq = tile / 13;
#pragma unroll
      for (int j = 0; j < 8; ++j) {
        int row = mtq * 16 + 8 * lh + j;
        buf[row * BUFSTRIDE + ntq * 16 + m16] = acc[q][j];
      }
    }
  }
  __syncthreads();

  // propagate + bias + relu -> write h_seq (bf16) in [B,N,T,H]
  if (tid < Hh) {
    float s = 0.0f;
    for (int n = 0; n < Nn; ++n) s += buf[n * BUFSTRIDE + tid];
    cs[tid] = s;
  }
  __syncthreads();
  for (int i = tid; i < Nn * Hh; i += 256) {
    int n = i / Hh, h = i % Hh;
    float v = (cs[h] + buf[n * BUFSTRIDE + h]) * inv + b2[h];
    v = v > 0.0f ? v : 0.0f;
    hseq[(((size_t)b * Nn + n) * Tt + t) * Hh + h] = (__bf16)v;
  }
}

// ---------------- LSTM single time step ----------------
// 207 blocks x 64 rows; 8 waves: wave = (colstrip<<2)|rowtile; gates via LDS.
__device__ __forceinline__ int gswz(int rl, int n) {          // bank-swizzled gate index
  return rl * G4 + (n ^ ((rl & 15) << 2));
}

__global__ __launch_bounds__(256) void lstm_step(
    const __bf16* __restrict__ hseq,  // [ROWS, T, H] bf16
    int t,
    const __bf16* __restrict__ hin,   // [ROWS, H] bf16
    __bf16* __restrict__ hout,        // [ROWS, H] bf16
    float* __restrict__ cst,          // [ROWS, H] f32 (in-place)
    const __bf16* __restrict__ wihbf, // [4H, H] bf16
    const __bf16* __restrict__ whhbf, // [4H, H] bf16
    const float* __restrict__ bih,    // [4H]
    const float* __restrict__ bhh)    // [4H]
{
  __shared__ float g[64 * G4];  // 64 KB gate exchange buffer

  const int tid = threadIdx.x, lane = tid & 31, wave = tid >> 5;
  const int lh = lane >> 4, m16 = lane & 15;
  const int r0 = blockIdx.x * 64;
  const int rt = wave & 3;             // 16-row tile within block
  const int csb = (wave >> 2) * 128;   // gate-column strip base (0 or 128)
  const int rowg = r0 + rt * 16 + m16;

  const __bf16* xrow = hseq + ((size_t)rowg * Tt + t) * Hh;
  const __bf16* hrow = hin + (size_t)rowg * Hh;

  v8f acc[8];
#pragma unroll
  for (int ct = 0; ct < 8; ++ct) {
    const int n = csb + ct * 16 + m16;
    const float bb = bih[n] + bhh[n];
    v8f z;
#pragma unroll
    for (int j = 0; j < 8; ++j) z[j] = bb;
    acc[ct] = z;
  }

#pragma unroll
  for (int kb = 0; kb < 2; ++kb) {
    v16bf Ax = loadA_bf16row(xrow, kb, lh);
    v16bf Ah = loadA_bf16row(hrow, kb, lh);
#pragma unroll
    for (int ct = 0; ct < 8; ++ct) {
      const int n = csb + ct * 16 + m16;
      v16bf Bi = loadB_bf16(wihbf, n, kb, lh);
      v16bf Bh = loadB_bf16(whhbf, n, kb, lh);
      acc[ct] = WMMA_BF16(Ax, Bi, acc[ct]);
      acc[ct] = WMMA_BF16(Ah, Bh, acc[ct]);
    }
  }

  // gates -> LDS
#pragma unroll
  for (int ct = 0; ct < 8; ++ct) {
#pragma unroll
    for (int j = 0; j < 8; ++j) {
      int rl = rt * 16 + 8 * lh + j;
      g[gswz(rl, csb + ct * 16 + m16)] = acc[ct][j];
    }
  }
  __syncthreads();

  // elementwise update: thread -> (row rl, 16 hidden units)
  const int rl = tid >> 2;
  const int h0 = (tid & 3) * 16;
  const int rg = r0 + rl;
#pragma unroll
  for (int j = 0; j < 16; ++j) {
    const int h = h0 + j;
    float gi = g[gswz(rl, h)];
    float gf = g[gswz(rl, 64 + h)];
    float gg = g[gswz(rl, 128 + h)];
    float go = g[gswz(rl, 192 + h)];
    float cp = cst[(size_t)rg * Hh + h];
    float si = 1.0f / (1.0f + __expf(-gi));
    float sf = 1.0f / (1.0f + __expf(-gf));
    float so = 1.0f / (1.0f + __expf(-go));
    float cn = sf * cp + si * tanhf(gg);
    cst[(size_t)rg * Hh + h] = cn;
    hout[(size_t)rg * Hh + h] = (__bf16)(so * tanhf(cn));
  }
}

// ---------------- output projection ----------------
__global__ __launch_bounds__(256) void fc_kernel(
    const __bf16* __restrict__ hT, const float* __restrict__ Wfc,
    const float* __restrict__ bfc, float* __restrict__ out)
{
  const int r = blockIdx.x * 256 + threadIdx.x;
  if (r >= ROWS) return;
  const __bf16* hp = hT + (size_t)r * Hh;
  float s = 0.0f;
#pragma unroll
  for (int k = 0; k < Hh; ++k) s += (float)hp[k] * Wfc[k];
  out[r] = s + bfc[0];
}

__global__ __launch_bounds__(256) void zero_kernel(float* p, size_t n) {
  size_t i = (size_t)blockIdx.x * 256 + threadIdx.x;
  if (i < n) p[i] = 0.0f;
}

// ---------------- host launcher ----------------
extern "C" void kernel_launch(void* const* d_in, const int* in_sizes, int n_in,
                              void* d_out, int out_size, void* d_ws, size_t ws_size,
                              hipStream_t stream) {
  (void)in_sizes; (void)n_in; (void)out_size; (void)ws_size;
  const float* x   = (const float*)d_in[0];
  // d_in[1], d_in[2] = N1, N2: structurally dead (sigmoid>0 -> all-ones adjacency)
  const float* W1  = (const float*)d_in[3];
  const float* b1  = (const float*)d_in[4];
  const float* W2  = (const float*)d_in[5];
  const float* b2  = (const float*)d_in[6];
  const float* Wih = (const float*)d_in[7];
  const float* Whh = (const float*)d_in[8];
  const float* bih = (const float*)d_in[9];
  const float* bhh = (const float*)d_in[10];
  const float* Wfc = (const float*)d_in[11];
  const float* bfc = (const float*)d_in[12];
  float* out = (float*)d_out;

  // ---- workspace layout (bytes) ----
  const size_t hseq_elems = (size_t)ROWS * Tt * Hh;  // bf16
  const size_t st_elems   = (size_t)ROWS * Hh;
  char* ws = (char*)d_ws;

  __bf16* hseq = (__bf16*)ws;                        ws += hseq_elems * 2;   // ~81 MB
  __bf16* hA   = (__bf16*)ws;                        ws += st_elems * 2;
  __bf16* hB   = (__bf16*)ws;                        ws += st_elems * 2;
  float*  cbuf = (float*)ws;                         ws += st_elems * 4;
  __bf16* wihbf = (__bf16*)ws;                       ws += (size_t)G4 * Hh * 2;
  __bf16* whhbf = (__bf16*)ws;                       ws += (size_t)G4 * Hh * 2;
  __bf16* w2t   = (__bf16*)ws;                       ws += (size_t)Hh * Hh * 2;

  // zero h0 (bf16 x2) + c0 (f32): contiguous region starting at hA
  {
    size_t nfloats = st_elems * 2;  // (2*st bf16 + st f32) bytes / 4
    zero_kernel<<<(unsigned)((nfloats + 255) / 256), 256, 0, stream>>>((float*)hA, nfloats);
  }
  prep_weights<<<(G4 * Hh + 255) / 256, 256, 0, stream>>>(Wih, Whh, W2, wihbf, whhbf, w2t);

  gcn_fused<<<Bb * Tt, 256, 0, stream>>>(x, W1, b1, w2t, b2, hseq);

  for (int t = 0; t < Tt; ++t) {
    __bf16* hin  = (t & 1) ? hB : hA;
    __bf16* hout = (t & 1) ? hA : hB;
    lstm_step<<<ROWS / 64, 256, 0, stream>>>(hseq, t, hin, hout, cbuf,
                                             wihbf, whhbf, bih, bhh);
  }
  // after t=47 (odd), final h lives in hA
  fc_kernel<<<(ROWS + 255) / 256, 256, 0, stream>>>(hA, Wfc, bfc, out);
}
